// MolecularEncoder_70815420776483
// MI455X (gfx1250) — matrix operand order repaired and checked by reading, not compiled, not run on hardware
//
#include <hip/hip_runtime.h>
#include <hip/hip_bf16.h>
#include <stdint.h>

#define N_NODES  100000
#define N_EDGES  200000
#define N_GRAPHS 4000
#define F_IN     44
#define HDIM     256
#define EDIM     512
#define NLAYERS  4

typedef __attribute__((ext_vector_type(16))) __bf16        v16bf;
typedef __attribute__((ext_vector_type(8)))  float         v8f;
typedef __attribute__((ext_vector_type(8)))  unsigned int  v8u;

// round-to-nearest-even f32 -> bf16
__device__ __forceinline__ unsigned short f2bf(float f) {
    unsigned int x = __builtin_bit_cast(unsigned int, f);
    x = (x + 0x7FFFu + ((x >> 16) & 1u)) >> 16;
    return (unsigned short)x;
}
__device__ __forceinline__ unsigned int pack_bf16(float a, float b) {
    return (unsigned int)f2bf(a) | ((unsigned int)f2bf(b) << 16);
}

// -------------------------------------------------------------------------
// Pre-swizzle f32 weights W[K][N] (row-major) into a contiguous-k wave32 WMMA bf16
// B-fragment layout: per 32x16 (KxN) tile, 32 lanes x 8 dwords;
// lane = half*16 + n_local, dword r holds K = half*16 + 2r, 2r+1.
// Each lane's fragment becomes one contiguous 32-byte read.
// -------------------------------------------------------------------------
__global__ void convert_weight(const float* __restrict__ W,
                               unsigned int* __restrict__ out, int K, int N) {
    int idx = blockIdx.x * blockDim.x + threadIdx.x;
    int total = (K >> 5) * (N >> 4) * 256;
    if (idx >= total) return;
    int r    = idx & 7;
    int lane = (idx >> 3) & 31;
    int tile = idx >> 8;               // kt * NT + nt
    int NT   = N >> 4;
    int kt   = tile / NT;
    int nt   = tile - kt * NT;
    int lm   = lane & 15;
    int half = lane >> 4;
    int n    = nt * 16 + lm;
    int k0   = kt * 32 + half * 16 + r * 2;
    out[idx] = pack_bf16(W[(size_t)k0 * N + n], W[(size_t)(k0 + 1) * N + n]);
}

// -------------------------------------------------------------------------
// h = relu(x @ W_atom + b_atom)   [N_NODES x 44] @ [44 x 256]
// -------------------------------------------------------------------------
__global__ __launch_bounds__(256) void atom_embed(
    const float* __restrict__ x, const float* __restrict__ W,
    const float* __restrict__ b, float* __restrict__ h) {
    __shared__ float xs[F_IN];
    int node = blockIdx.x;
    if (threadIdx.x < F_IN) xs[threadIdx.x] = x[(size_t)node * F_IN + threadIdx.x];
    __syncthreads();
    int c = threadIdx.x;
    float acc = b[c];
#pragma unroll 4
    for (int k = 0; k < F_IN; ++k) acc = fmaf(xs[k], W[k * HDIM + c], acc);
    h[(size_t)node * HDIM + c] = fmaxf(acc, 0.0f);
}

// -------------------------------------------------------------------------
// agg[dst] += h[src]   (agg pre-initialized to h => z = h + sum_j h_j)
// -------------------------------------------------------------------------
__global__ __launch_bounds__(256) void scatter_add(
    const float* __restrict__ h, const int* __restrict__ ei,
    float* __restrict__ agg) {
    int t = blockIdx.x * blockDim.x + threadIdx.x;
    if (t >= N_EDGES * (HDIM / 4)) return;
    int e = t >> 6;
    int c = (t & 63) << 2;
    int s = ei[e];
    int d = ei[N_EDGES + e];
    float4 v = *reinterpret_cast<const float4*>(h + (size_t)s * HDIM + c);
    float* p = agg + (size_t)d * HDIM + c;
    atomicAdd(p + 0, v.x); atomicAdd(p + 1, v.y);
    atomicAdd(p + 2, v.z); atomicAdd(p + 3, v.w);
}

// -------------------------------------------------------------------------
// Mean-pool helpers
// -------------------------------------------------------------------------
__global__ __launch_bounds__(256) void pool_sum(
    const float* __restrict__ h, const int* __restrict__ batch,
    float* __restrict__ sums) {
    int t = blockIdx.x * blockDim.x + threadIdx.x;
    if (t >= N_NODES * (HDIM / 4)) return;
    int n = t >> 6;
    int c = (t & 63) << 2;
    int g = batch[n];
    float4 v = *reinterpret_cast<const float4*>(h + (size_t)n * HDIM + c);
    float* p = sums + (size_t)g * HDIM + c;
    atomicAdd(p + 0, v.x); atomicAdd(p + 1, v.y);
    atomicAdd(p + 2, v.z); atomicAdd(p + 3, v.w);
}

__global__ __launch_bounds__(256) void pool_count(
    const int* __restrict__ batch, float* __restrict__ counts) {
    int n = blockIdx.x * blockDim.x + threadIdx.x;
    if (n >= N_NODES) return;
    atomicAdd(&counts[batch[n]], 1.0f);
}

__global__ __launch_bounds__(256) void pool_div(
    const float* __restrict__ sums, const float* __restrict__ counts,
    float* __restrict__ hg) {
    int t = blockIdx.x * blockDim.x + threadIdx.x;
    if (t >= N_GRAPHS * HDIM) return;
    int g = t >> 8;
    hg[t] = sums[t] / fmaxf(counts[g], 1.0f);
}

// -------------------------------------------------------------------------
// C = [relu](A @ B + bias).  128x64 output tile, 256 threads (8 waves),
// each wave owns a 2x2 grid of 16x16 C tiles -> 4 v_wmma per 32-K step.
//   A_BF16 == 0: A is f32 [M x K]; staged through LDS with f32->bf16 convert.
//   A_BF16 == 1: A is bf16 [M x K]; staged with GLOBAL_LOAD_ASYNC_TO_LDS_B128
//                (ASYNCcnt) — pure DMA, no VGPR round trip.
//   OUT_BF16 selects bf16 (packed) or f32 output.
// B is pre-swizzled bf16 fragments (32B contiguous per lane per tile).
// -------------------------------------------------------------------------
template <int A_BF16, int OUT_BF16>
__global__ __launch_bounds__(256) void gemm_wmma(
    const void* __restrict__ Ap, const unsigned int* __restrict__ Bsw,
    const float* __restrict__ bias, void* __restrict__ Cp,
    int M, int K, int N, int doRelu) {
    __shared__ __align__(16) unsigned short lds_a[128 * 32];   // 8 KB bf16 tile

    const int tid  = threadIdx.x;
    const int wave = tid >> 5;
    const int lane = tid & 31;
    const int lm   = lane & 15;
    const int half = lane >> 4;
    const int miB  = (wave & 3) << 1;     // m-tiles miB, miB+1
    const int njB  = (wave >> 2) << 1;    // n-tiles njB, njB+1
    const int m0   = blockIdx.x * 128;
    const int n0   = blockIdx.y * 64;
    const int NT   = N >> 4;

    v8f acc00 = {}, acc01 = {}, acc10 = {}, acc11 = {};

    const int  srow  = tid >> 1;          // staging row 0..127
    const int  grow  = m0 + srow;
    const bool rowok = grow < M;
    const unsigned ldsbase = (unsigned)(size_t)(&lds_a[0]);

    for (int kk = 0; kk < K; kk += 32) {
        __syncthreads();                              // prior reads retired

        if constexpr (A_BF16 == 0) {
            // stage 128x32 f32 -> bf16 into LDS (16 f32 per thread)
            const int scol = (tid & 1) << 4;          // 0 or 16 f32 elements
            float4 f0 = {}, f1 = {}, f2 = {}, f3 = {};
            if (rowok) {
                const float* ar = (const float*)Ap + (size_t)grow * K + kk + scol;
                f0 = *reinterpret_cast<const float4*>(ar + 0);
                f1 = *reinterpret_cast<const float4*>(ar + 4);
                f2 = *reinterpret_cast<const float4*>(ar + 8);
                f3 = *reinterpret_cast<const float4*>(ar + 12);
            }
            uint4 p0, p1;
            p0.x = pack_bf16(f0.x, f0.y); p0.y = pack_bf16(f0.z, f0.w);
            p0.z = pack_bf16(f1.x, f1.y); p0.w = pack_bf16(f1.z, f1.w);
            p1.x = pack_bf16(f2.x, f2.y); p1.y = pack_bf16(f2.z, f2.w);
            p1.z = pack_bf16(f3.x, f3.y); p1.w = pack_bf16(f3.z, f3.w);
            uint4* dst = reinterpret_cast<uint4*>(&lds_a[srow * 32 + scol]);
            dst[0] = p0;
            dst[1] = p1;
        } else {
            // async DMA: 32 bytes per thread, global -> LDS, tracked by ASYNCcnt
            if (rowok) {
                const int colb = (tid & 1) << 5;      // 0 or 32 bytes within row
                unsigned long long g =
                    (unsigned long long)((const char*)Ap +
                        (((size_t)grow * K + kk) << 1) + colb);
                unsigned loff = ldsbase + (unsigned)(srow * 64 + colb);
                asm volatile("global_load_async_to_lds_b128 %0, %1, off"
                             :: "v"(loff), "v"(g) : "memory");
                asm volatile("global_load_async_to_lds_b128 %0, %1, off offset:16"
                             :: "v"(loff), "v"(g) : "memory");
            }
            asm volatile("s_wait_asynccnt 0" ::: "memory");
        }
        __syncthreads();                              // tile visible to all waves

        // ---- A fragments (16-bit A 16x32, contiguous k order) ----
        // lanes 0-15: row=lm, K 0-15; lanes 16-31: K 16-31 (matches B swizzle)
        v16bf af[2];
#pragma unroll
        for (int mi = 0; mi < 2; ++mi) {
            const unsigned short* ab =
                &lds_a[((miB + mi) * 16 + lm) * 32 + half * 16];
            uint4 a0 = *reinterpret_cast<const uint4*>(ab);
            uint4 a1 = *reinterpret_cast<const uint4*>(ab + 8);
            v8u au;
            au[0] = a0.x; au[1] = a0.y; au[2] = a0.z; au[3] = a0.w;
            au[4] = a1.x; au[5] = a1.y; au[6] = a1.z; au[7] = a1.w;
            af[mi] = __builtin_bit_cast(v16bf, au);
        }

        // ---- B fragments (contiguous 32B per lane, pre-swizzled) ----
        const int kt  = kk >> 5;
        const int ntg = (n0 >> 4) + njB;
        v16bf bfm[2];
#pragma unroll
        for (int nj = 0; nj < 2; ++nj) {
            const uint4* bp = reinterpret_cast<const uint4*>(
                Bsw + ((size_t)(kt * NT + ntg + nj) * 32 + lane) * 8);
            uint4 b0 = bp[0], b1 = bp[1];
            v8u bu;
            bu[0] = b0.x; bu[1] = b0.y; bu[2] = b0.z; bu[3] = b0.w;
            bu[4] = b1.x; bu[5] = b1.y; bu[6] = b1.z; bu[7] = b1.w;
            bfm[nj] = __builtin_bit_cast(v16bf, bu);
        }

        acc00 = __builtin_amdgcn_wmma_f32_16x16x32_bf16(
            false, af[0], false, bfm[0], (short)0, acc00, false, false);
        acc01 = __builtin_amdgcn_wmma_f32_16x16x32_bf16(
            false, af[0], false, bfm[1], (short)0, acc01, false, false);
        acc10 = __builtin_amdgcn_wmma_f32_16x16x32_bf16(
            false, af[1], false, bfm[0], (short)0, acc10, false, false);
        acc11 = __builtin_amdgcn_wmma_f32_16x16x32_bf16(
            false, af[1], false, bfm[1], (short)0, acc11, false, false);
    }

    // ---- epilogue: C vgpr r -> m_local = half*8 + r, n_local = lm ----
    auto store_tile = [&](int mi, int nj, const v8f& a) {
        int n = n0 + (njB + nj) * 16 + lm;
        float bv = bias[n];
        int mrow = m0 + (miB + mi) * 16 + half * 8;
#pragma unroll
        for (int r = 0; r < 8; ++r) {
            int m = mrow + r;
            if (m < M) {
                float v = a[r] + bv;
                if (doRelu) v = fmaxf(v, 0.0f);
                if constexpr (OUT_BF16)
                    ((unsigned short*)Cp)[(size_t)m * N + n] = f2bf(v);
                else
                    ((float*)Cp)[(size_t)m * N + n] = v;
            }
        }
    };
    store_tile(0, 0, acc00);
    store_tile(0, 1, acc01);
    store_tile(1, 0, acc10);
    store_tile(1, 1, acc11);
}

// -------------------------------------------------------------------------
extern "C" void kernel_launch(void* const* d_in, const int* in_sizes, int n_in,
                              void* d_out, int out_size, void* d_ws, size_t ws_size,
                              hipStream_t stream) {
    (void)in_sizes; (void)n_in; (void)out_size; (void)ws_size;

    const float* x      = (const float*)d_in[0];
    const int*   ei     = (const int*)d_in[1];
    const int*   batch  = (const int*)d_in[2];
    const float* W_atom = (const float*)d_in[3];
    const float* b_atom = (const float*)d_in[4];
    const float* W1     = (const float*)d_in[5];
    const float* b1     = (const float*)d_in[6];
    const float* W2     = (const float*)d_in[7];
    const float* b2     = (const float*)d_in[8];
    const float* Wf1    = (const float*)d_in[9];
    const float* bf1    = (const float*)d_in[10];
    const float* Wf2    = (const float*)d_in[11];
    const float* bf2    = (const float*)d_in[12];
    float* out = (float*)d_out;

    // ---- workspace carve-up (256B aligned) ----
    char* ws = (char*)d_ws;
    size_t off = 0;
    auto carve = [&](size_t bytes) -> char* {
        char* p = ws + off;
        off += (bytes + 255) & ~(size_t)255;
        return p;
    };
    float*          h    = (float*)carve((size_t)N_NODES * HDIM * 4);   // 102.4 MB
    float*          zin  = (float*)carve((size_t)N_NODES * HDIM * 4);   // 102.4 MB
    unsigned short* zmid = (unsigned short*)carve((size_t)N_NODES * EDIM * 2); // 102.4 MB bf16
    float*          sums = (float*)carve((size_t)N_GRAPHS * HDIM * 4);
    float*          cnts = (float*)carve((size_t)N_GRAPHS * 4);
    float*          hg   = (float*)carve((size_t)N_GRAPHS * HDIM * 4);
    unsigned short* t1   = (unsigned short*)carve((size_t)N_GRAPHS * HDIM * 2);
    const int cw1  = (HDIM / 32) * (2 * HDIM / 16) * 256;   // dwords per W1 layer
    const int cw2  = (2 * HDIM / 32) * (HDIM / 16) * 256;   // dwords per W2 layer
    const int cwf1 = (HDIM / 32) * (HDIM / 16) * 256;
    const int cwf2 = (HDIM / 32) * (EDIM / 16) * 256;
    unsigned int* w1sw  = (unsigned int*)carve((size_t)NLAYERS * cw1 * 4);
    unsigned int* w2sw  = (unsigned int*)carve((size_t)NLAYERS * cw2 * 4);
    unsigned int* wf1sw = (unsigned int*)carve((size_t)cwf1 * 4);
    unsigned int* wf2sw = (unsigned int*)carve((size_t)cwf2 * 4);

    // ---- weight swizzle to bf16 fragment layout ----
    for (int l = 0; l < NLAYERS; ++l)
        convert_weight<<<(cw1 + 255) / 256, 256, 0, stream>>>(
            W1 + (size_t)l * HDIM * 2 * HDIM, w1sw + (size_t)l * cw1, HDIM, 2 * HDIM);
    for (int l = 0; l < NLAYERS; ++l)
        convert_weight<<<(cw2 + 255) / 256, 256, 0, stream>>>(
            W2 + (size_t)l * 2 * HDIM * HDIM, w2sw + (size_t)l * cw2, 2 * HDIM, HDIM);
    convert_weight<<<(cwf1 + 255) / 256, 256, 0, stream>>>(Wf1, wf1sw, HDIM, HDIM);
    convert_weight<<<(cwf2 + 255) / 256, 256, 0, stream>>>(Wf2, wf2sw, HDIM, EDIM);

    // ---- atom embedding ----
    atom_embed<<<N_NODES, 256, 0, stream>>>(x, W_atom, b_atom, h);

    // ---- GIN layers ----
    for (int l = 0; l < NLAYERS; ++l) {
        hipMemcpyAsync(zin, h, (size_t)N_NODES * HDIM * 4,
                       hipMemcpyDeviceToDevice, stream);
        scatter_add<<<(N_EDGES * (HDIM / 4) + 255) / 256, 256, 0, stream>>>(h, ei, zin);
        dim3 g1((N_NODES + 127) / 128, (2 * HDIM) / 64);
        gemm_wmma<0, 1><<<g1, 256, 0, stream>>>(
            zin, w1sw + (size_t)l * cw1, b1 + (size_t)l * 2 * HDIM, zmid,
            N_NODES, HDIM, 2 * HDIM, 1);
        dim3 g2((N_NODES + 127) / 128, HDIM / 64);
        gemm_wmma<1, 0><<<g2, 256, 0, stream>>>(
            zmid, w2sw + (size_t)l * cw2, b2 + (size_t)l * HDIM, h,
            N_NODES, 2 * HDIM, HDIM, 1);
    }

    // ---- global mean pool ----
    hipMemsetAsync(sums, 0, (size_t)N_GRAPHS * HDIM * 4, stream);
    hipMemsetAsync(cnts, 0, (size_t)N_GRAPHS * 4, stream);
    pool_sum<<<(N_NODES * (HDIM / 4) + 255) / 256, 256, 0, stream>>>(h, batch, sums);
    pool_count<<<(N_NODES + 255) / 256, 256, 0, stream>>>(batch, cnts);
    pool_div<<<(N_GRAPHS * HDIM + 255) / 256, 256, 0, stream>>>(sums, cnts, hg);

    // ---- graph MLP head ----
    dim3 gf1((N_GRAPHS + 127) / 128, HDIM / 64);
    gemm_wmma<0, 1><<<gf1, 256, 0, stream>>>(hg, wf1sw, bf1, t1,
                                             N_GRAPHS, HDIM, HDIM, 1);
    dim3 gf2((N_GRAPHS + 127) / 128, EDIM / 64);
    gemm_wmma<1, 0><<<gf2, 256, 0, stream>>>(t1, wf2sw, bf2, out,
                                             N_GRAPHS, HDIM, EDIM, 0);
}
